// ExecutionUnit_35235911696734
// MI455X (gfx1250) — matrix operand order repaired and checked by reading, not compiled, not run on hardware
//
#include <hip/hip_runtime.h>
#include <hip/hip_bf16.h>

// Problem constants (from reference): B=64, T=64, P=16, D=256
#define B_  64
#define T_  64
#define P_  16
#define D_  256
#define LDP 264            // padded LDS stride in halves (132 dwords -> bank-conflict-free)
#define MROWS 128          // rows of `action` per workgroup (2 WGs per batch)
#define NTH 1024           // 32 waves: 4 C-tiles/wave -> low VGPR pressure

typedef __attribute__((ext_vector_type(16))) _Float16 v16h;
typedef __attribute__((ext_vector_type(8)))  _Float16 v8h;
typedef __attribute__((ext_vector_type(2)))  _Float16 h2;
typedef __attribute__((ext_vector_type(8)))  float    v8f;

// ---------------------------------------------------------------------------
// Kernel 1: prims[0] = I, prims[p>0] = row-softmax(scores[p-1]).
// Stored TRANSPOSED in f16: primsT[p][j][i] = prims[p][i][j]  (j = softmax axis)
// so kernel 2 reads WMMA B-operand columns with contiguous 128-bit loads.
// ---------------------------------------------------------------------------
__global__ __launch_bounds__(256) void prims_softmax_kernel(
    const float* __restrict__ scores, _Float16* __restrict__ primsT) {
  const int blk = blockIdx.x;          // p*256 + i
  const int p   = blk >> 8;
  const int i   = blk & 255;
  const int j   = threadIdx.x;
  __shared__ float red[256];

  float y;
  if (p == 0) {
    y = (i == j) ? 1.0f : 0.0f;        // identity primitive
  } else {
    const float x = scores[(((size_t)(p - 1)) * D_ + i) * D_ + j];
    red[j] = x; __syncthreads();
    for (int s = 128; s > 0; s >>= 1) {
      if (j < s) red[j] = fmaxf(red[j], red[j + s]);
      __syncthreads();
    }
    const float mx = red[0]; __syncthreads();
    const float e = __expf(x - mx);
    red[j] = e; __syncthreads();
    for (int s = 128; s > 0; s >>= 1) {
      if (j < s) red[j] += red[j + s];
      __syncthreads();
    }
    y = e / red[0];
  }
  primsT[(size_t)p * D_ * D_ + (size_t)j * D_ + i] = (_Float16)y;
}

// ---------------------------------------------------------------------------
// Kernel 2: per-batch chain of T matmuls, WMMA f32 <- f16.
//   grid.x = 128 (2 WGs per batch, 128 action rows each), 1024 threads.
//   LDS: S (column-major f16, 256x264) + double-buffered Act (2 x 128x264)
//   = 264 KB. Two barriers per step. Wave handles one 16-row strip x 64 cols:
//   per K-block, ALL four B fragments are loaded as live values before the
//   four WMMAs (staircase dscnt waits), and sched_group_barrier pins the
//   DS-read/WMMA grouping so the scheduler can't re-serialize it.
// ---------------------------------------------------------------------------
__global__ __launch_bounds__(NTH) void chain_kernel(
    const float* __restrict__ control,
    const _Float16* __restrict__ primsT,
    float* __restrict__ out) {
  extern __shared__ char smem[];
  _Float16* __restrict__ Slds = (_Float16*)smem;                           // D_*LDP
  _Float16* __restrict__ A0   = (_Float16*)(smem + (size_t)D_ * LDP * 2);  // MROWS*LDP
  _Float16* __restrict__ A1   = A0 + (size_t)MROWS * LDP;                  // MROWS*LDP

  const int tid  = threadIdx.x;
  const int b    = blockIdx.x >> 1;
  const int mblk = blockIdx.x & 1;
  const int m0   = mblk * MROWS;

  const int lane = tid & 31;
  const int wv   = tid >> 5;       // wave 0..31
  const int rt   = wv & 7;         // row tile (16 rows)
  const int cq   = wv >> 3;        // col quarter (64 cols)
  const int R    = rt * 16;        // local row base
  const int C0   = cq * 64;        // local col base
  const int ln16 = lane & 15;
  const int hi   = lane >> 4;      // lane half (per ISA fragment layout)

  // A-fragment loader (16x32 f16 per ISA: lanes0-15 K {0-7,16-23}, lanes16-31 K {8-15,24-31})
  auto loadA = [&](const _Float16* __restrict__ Ab, int kb) -> v16h {
    const _Float16* abase = Ab + (size_t)(R + ln16) * LDP + kb * 32;
    v16h a;
    *((v8h*)&a)       = *(const v8h*)(abase + (hi ? 8 : 0));
    *(((v8h*)&a) + 1) = *(const v8h*)(abase + 16 + (hi ? 8 : 0));
    return a;
  };
  // B-fragment loader (32x16 f16: column per lane; lanes0-15 K {0-7,16-23}, lanes16-31 K {8-15,24-31})
  auto loadB = [&](int kb, int c) -> v16h {
    const _Float16* bbase =
        Slds + (size_t)(C0 + c * 16 + ln16) * LDP + kb * 32 + (hi ? 8 : 0);
    v16h bb;
    *((v8h*)&bb)       = *(const v8h*)(bbase);
    *(((v8h*)&bb) + 1) = *(const v8h*)(bbase + 16);
    return bb;
  };

  // Act[buf0] := identity rows m0..m0+127 (pad region zeroed)
  for (int idx = tid; idx < MROWS * LDP; idx += NTH) {
    const int m = idx / LDP, k = idx % LDP;
    A0[idx] = (_Float16)((k < D_ && (m0 + m) == k) ? 1.0f : 0.0f);
  }

  v8f acc[4];

  #pragma unroll 1
  for (int t = 0; t < T_; ++t) {
    const _Float16* __restrict__ Acur = (t & 1) ? A1 : A0;
    _Float16* __restrict__       Anxt = (t & 1) ? A0 : A1;

    // ---- per-step control weights (uniform -> scalar loads) ----
    const float* __restrict__ cw = control + ((size_t)b * T_ + t) * P_;
    h2 w2[P_];
    #pragma unroll
    for (int p = 0; p < P_; ++p) {
      const _Float16 wh = (_Float16)cw[p];
      h2 v; v[0] = wh; v[1] = wh;
      w2[p] = v;
    }

    // ---- build S_t column-major in LDS via packed-f16 FMA ----
    // (prev step's S reads retired at the end-of-step barrier)
    #pragma unroll
    for (int v = 0; v < 8; ++v) {
      const int g  = v * NTH + tid;      // 0..8191 vectors of 8 halves
      const int j  = g >> 5;             // column (N index)
      const int k0 = (g & 31) * 8;       // K base
      h2 s[4];
      #pragma unroll
      for (int q = 0; q < 4; ++q) { h2 z; z[0] = (_Float16)0; z[1] = (_Float16)0; s[q] = z; }
      #pragma unroll
      for (int p = 0; p < P_; ++p) {
        const v8h pv = *(const v8h*)(primsT + (size_t)p * D_ * D_ + (size_t)j * D_ + k0);
        const h2* pc = (const h2*)&pv;
        #pragma unroll
        for (int q = 0; q < 4; ++q) s[q] += w2[p] * pc[q];   // v_pk_fma_f16
      }
      *(v8h*)(Slds + (size_t)j * LDP + k0) = *(const v8h*)&s[0];
    }
    __syncthreads();  // S_t (and Act writes from prev step / identity init) visible

    // ---- WMMA phase: acc = Act_rows @ S_t ----
    #pragma unroll
    for (int c = 0; c < 4; ++c) {
      v8f z = {0.f, 0.f, 0.f, 0.f, 0.f, 0.f, 0.f, 0.f};
      acc[c] = z;
    }
    #pragma unroll
    for (int kb = 0; kb < 8; ++kb) {
      // All 10 ds_loads issued before any WMMA of this K-block: distinct,
      // simultaneously-live fragments -> RA cannot coalesce, waits staircase.
      v16h a  = loadA(Acur, kb);
      v16h b0 = loadB(kb, 0);
      v16h b1 = loadB(kb, 1);
      v16h b2 = loadB(kb, 2);
      v16h b3 = loadB(kb, 3);
      acc[0] = __builtin_amdgcn_wmma_f32_16x16x32_f16(false, a, false, b0,
                                                      (short)0, acc[0], false, false);
      acc[1] = __builtin_amdgcn_wmma_f32_16x16x32_f16(false, a, false, b1,
                                                      (short)0, acc[1], false, false);
      acc[2] = __builtin_amdgcn_wmma_f32_16x16x32_f16(false, a, false, b2,
                                                      (short)0, acc[2], false, false);
      acc[3] = __builtin_amdgcn_wmma_f32_16x16x32_f16(false, a, false, b3,
                                                      (short)0, acc[3], false, false);
#if __has_builtin(__builtin_amdgcn_sched_group_barrier)
      // Pin grouping: 10 DS reads, then 4 matrix ops, per K-block.
      __builtin_amdgcn_sched_group_barrier(0x100, 10, 0);  // DS_READ
      __builtin_amdgcn_sched_group_barrier(0x008, 4, 0);   // MFMA/WMMA
#endif
    }

    // ---- write back: C tile VGPR rr -> (M = rr + 8*hi, N = ln16) ----
    if (t < T_ - 1) {
      #pragma unroll
      for (int c = 0; c < 4; ++c) {
        const int n = C0 + c * 16 + ln16;
        #pragma unroll
        for (int rr = 0; rr < 8; ++rr) {
          const int m = R + rr + hi * 8;
          Anxt[(size_t)m * LDP + n] = (_Float16)acc[c][rr];
        }
      }
    } else {
      float* __restrict__ ob = out + (size_t)b * D_ * D_;
      #pragma unroll
      for (int c = 0; c < 4; ++c) {
        const int n = C0 + c * 16 + ln16;
        #pragma unroll
        for (int rr = 0; rr < 8; ++rr) {
          const int m = m0 + R + rr + hi * 8;
          ob[(size_t)m * D_ + n] = acc[c][rr];
        }
      }
    }
    __syncthreads();  // WMMA reads of S/Acur done; Anxt writes visible next step
  }
}

// ---------------------------------------------------------------------------
extern "C" void kernel_launch(void* const* d_in, const int* in_sizes, int n_in,
                              void* d_out, int out_size, void* d_ws, size_t ws_size,
                              hipStream_t stream) {
  (void)in_sizes; (void)n_in; (void)out_size; (void)ws_size;
  const float* control = (const float*)d_in[0];           // [B,T,P] f32
  const float* scores  = (const float*)d_in[1];           // [P-1,D,D] f32
  float* out           = (float*)d_out;                   // [B,D,D] f32
  _Float16* primsT     = (_Float16*)d_ws;                 // [P,D,D] f16 transposed (2 MB)

  prims_softmax_kernel<<<P_ * D_, 256, 0, stream>>>(scores, primsT);

  const size_t lds_bytes =
      ((size_t)D_ * LDP + 2 * (size_t)MROWS * LDP) * sizeof(_Float16);  // ~264 KB
  chain_kernel<<<B_ * 2, NTH, lds_bytes, stream>>>(control, primsT, out);
}
